// SelfAttention_59605556134455
// MI455X (gfx1250) — matrix operand order repaired and checked by reading, not compiled, not run on hardware
//
#include <hip/hip_runtime.h>

// ---------------------------------------------------------------------------
// Self-attention, B=4 S=2048 D=1024, fp32 in/out.
//   Kernel 0: fp32 -> bf16 conversion pre-pass for x, Wq, Wk, Wv (mem-bound).
//   Kernel 1: Q/K = x*W^T (row-major bf16), V stored TRANSPOSED Vt[B][D][S].
//             Pure-bf16 GEMM, 32x64 tile per wave, 8 WMMAs per k-step.
//   Kernel 2: causal flash attention, one block per (batch, 16-query tile),
//             head dim split across 8 waves (128 cols each).
// ---------------------------------------------------------------------------

typedef __attribute__((ext_vector_type(16))) __bf16 v16bf;
typedef __attribute__((ext_vector_type(8)))  __bf16 v8bf;
typedef __attribute__((ext_vector_type(8)))  float  v8f;
typedef __attribute__((ext_vector_type(8)))  float  f32x8;

#define BATCH 4
#define SEQ   2048
#define DIM   1024
#define SCALE 0.03125f   // 1/sqrt(1024)

__device__ __forceinline__ v16bf join88(v8bf lo, v8bf hh) {
  v16bf r;
#pragma unroll
  for (int i = 0; i < 8; ++i) { r[i] = lo[i]; r[i + 8] = hh[i]; }
  return r;
}

__device__ __forceinline__ v8f wmma_bf16(v16bf a, v16bf b, v8f c) {
  return __builtin_amdgcn_wmma_f32_16x16x32_bf16(
      /*neg_a=*/false, a, /*neg_b=*/false, b,
      /*c_mod=*/(short)0, c, /*reuse_a=*/false, /*reuse_b=*/false);
}

// ---------------------------------------------------------------------------
// Kernel 0: fp32 -> bf16 conversion. blockIdx.y selects source array.
// ---------------------------------------------------------------------------
__global__ __launch_bounds__(256) void cvt_bf16_kernel(
    const float* __restrict__ s0, const float* __restrict__ s1,
    const float* __restrict__ s2, const float* __restrict__ s3,
    __bf16* __restrict__ d0, __bf16* __restrict__ d1,
    __bf16* __restrict__ d2, __bf16* __restrict__ d3)
{
  const int y = blockIdx.y;
  const float* s = (y == 0) ? s0 : (y == 1) ? s1 : (y == 2) ? s2 : s3;
  __bf16*      d = (y == 0) ? d0 : (y == 1) ? d1 : (y == 2) ? d2 : d3;
  const size_t count = (y == 0) ? (size_t)BATCH * SEQ * DIM : (size_t)DIM * DIM;
  const size_t i = ((size_t)blockIdx.x * 256 + threadIdx.x) * 8;
  if (i >= count) return;
  const f32x8 f = *(const f32x8*)(s + i);
  *(v8bf*)(d + i) = __builtin_convertvector(f, v8bf);  // v_cvt_pk_bf16_f32
}

// ---------------------------------------------------------------------------
// Kernel 1: QKV projection, pure-bf16 inputs.  Each wave owns one 32x64 tile
// of one output matrix.  3 * (8192/32) * (1024/64) = 12288 wave-tiles,
// 8 waves/block -> 1536 blocks.
// ---------------------------------------------------------------------------
__global__ __launch_bounds__(256) void qkv_proj_kernel(
    const __bf16* __restrict__ xb,   // [B*S][D] bf16
    const __bf16* __restrict__ Wb,   // [3][D][D] bf16 (Wq,Wk,Wv)
    __bf16* __restrict__ Qb,         // [B*S][D]
    __bf16* __restrict__ Kb,         // [B*S][D]
    __bf16* __restrict__ Vt)         // [B][D][S] (transposed)
{
  const int wave = threadIdx.x >> 5;
  const int lane = threadIdx.x & 31;
  const int t    = blockIdx.x * 8 + wave;
  const int mat  = t >> 12;            // 0=Q, 1=K, 2=V  (4096 tiles per mat)
  const int rem  = t & 4095;
  const int row0 = (rem >> 4) << 5;    // 32-row tile over B*S
  const int n0   = (rem & 15) << 6;    // 64-col chunk
  const int l16  = lane & 15;
  const int hi   = lane >> 4;
  const int kb8  = hi << 3;
  const int kb16 = hi << 4;

  const __bf16* W   = Wb + (size_t)mat * DIM * DIM;
  const __bf16* xr0 = xb + (size_t)(row0 + l16) * DIM + kb16;
  const __bf16* xr1 = xr0 + (size_t)16 * DIM;

  const v8f vz = {0.f, 0.f, 0.f, 0.f, 0.f, 0.f, 0.f, 0.f};
  v8f acc[8];
#pragma unroll
  for (int c = 0; c < 8; ++c) acc[c] = vz;

  for (int kk = 0; kk < DIM; kk += 32) {
    const v16bf aF0 = join88(*(const v8bf*)(xr0 + kk),
                             *(const v8bf*)(xr0 + kk + 8));
    const v16bf aF1 = join88(*(const v8bf*)(xr1 + kk),
                             *(const v8bf*)(xr1 + kk + 8));
#pragma unroll
    for (int c = 0; c < 4; ++c) {
      const v16bf bF =
          *(const v16bf*)(W + (size_t)(n0 + (c << 4) + l16) * DIM + kk + kb16);
      acc[c]     = wmma_bf16(aF0, bF, acc[c]);
      acc[4 + c] = wmma_bf16(aF1, bF, acc[4 + c]);
    }
  }

  if (mat < 2) {
    __bf16* dst = mat ? Kb : Qb;
#pragma unroll
    for (int h = 0; h < 2; ++h) {
#pragma unroll
      for (int c = 0; c < 4; ++c) {
        const v8bf vb = __builtin_convertvector(acc[h * 4 + c], v8bf);
        const int n = n0 + (c << 4) + l16;
#pragma unroll
        for (int r = 0; r < 8; ++r)
          dst[(size_t)(row0 + h * 16 + r + (hi << 3)) * DIM + n] = vb[r];
      }
    }
  } else {
    // V transposed: Vt[b][d][s]; 8 consecutive s per lane -> one 16B store
    const int bb = row0 >> 11;
#pragma unroll
    for (int h = 0; h < 2; ++h) {
      const int s0 = (row0 & 2047) + h * 16 + (hi << 3);
#pragma unroll
      for (int c = 0; c < 4; ++c) {
        const int d = n0 + (c << 4) + l16;
        *(v8bf*)(Vt + (size_t)(bb * DIM + d) * SEQ + s0) =
            __builtin_convertvector(acc[h * 4 + c], v8bf);
      }
    }
  }
}

// ---------------------------------------------------------------------------
// Kernel 2: causal flash attention. Grid = (S/16, B), 256 threads (8 waves).
// Wave w accumulates output columns [128w, 128w+128).
// ---------------------------------------------------------------------------
__global__ __launch_bounds__(256) void attn_kernel(
    const __bf16* __restrict__ Qb,
    const __bf16* __restrict__ Kb,
    const __bf16* __restrict__ Vt,
    float* __restrict__ out)
{
  __shared__ __align__(16) float  S_lds[16][32];   // score accumulator
  __shared__ __align__(16) __bf16 P_lds[16][32];   // softmax probs
  __shared__ float alpha_l[16];
  __shared__ float l_l[16];

  const int b     = blockIdx.y;
  const int q0    = blockIdx.x << 4;
  const int wave  = threadIdx.x >> 5;
  const int lane  = threadIdx.x & 31;
  const int l16   = lane & 15;
  const int hi    = lane >> 4;
  const int kb8   = hi << 3;
  const int kb16  = hi << 4;
  const int dbase = wave << 7;
  const int bS    = b * SEQ;

  // Preload Q A-fragments for this wave's 128-wide D slice (loop-invariant).
  v16bf aQ[4];
  {
    const __bf16* qrow = Qb + (size_t)(bS + q0 + l16) * DIM + dbase + kb16;
#pragma unroll
    for (int c = 0; c < 4; ++c) {
      aQ[c] = join88(*(const v8bf*)(qrow + c * 32),
                     *(const v8bf*)(qrow + c * 32 + 8));
    }
  }

  const v8f vz = {0.f, 0.f, 0.f, 0.f, 0.f, 0.f, 0.f, 0.f};
  v8f O[8];
#pragma unroll
  for (int f = 0; f < 8; ++f) O[f] = vz;

  float m_run = -1e30f, l_run = 0.f;   // live on wave 0, lanes<16

  const int jmax = (q0 + 15) >> 5;
  for (int j = 0; j <= jmax; ++j) {
    // -- zero score accumulator ------------------------------------------
    {
      float* S1 = &S_lds[0][0];
      S1[threadIdx.x]       = 0.f;
      S1[threadIdx.x + 256] = 0.f;
    }
    __syncthreads();

    // -- partial S = Q K^T over this wave's D slice ----------------------
    const int key0 = j << 5;
    v8f acc0 = vz, acc1 = vz;
#pragma unroll
    for (int c = 0; c < 4; ++c) {
      const __bf16* kp =
          Kb + (size_t)(bS + key0 + l16) * DIM + dbase + c * 32 + kb16;
      const v16bf bK0 = *(const v16bf*)(kp);
      const v16bf bK1 = *(const v16bf*)(kp + (size_t)16 * DIM);
      acc0 = wmma_bf16(aQ[c], bK0, acc0);
      acc1 = wmma_bf16(aQ[c], bK1, acc1);
    }
#pragma unroll
    for (int r = 0; r < 8; ++r) {
      atomicAdd(&S_lds[r + (hi << 3)][l16],      acc0[r]);
      atomicAdd(&S_lds[r + (hi << 3)][l16 + 16], acc1[r]);
    }
    __syncthreads();

    // -- online softmax (wave 0, lanes 0..15, one row each) --------------
    if (wave == 0 && lane < 16) {
      const int row = lane;
      const int q   = q0 + row;
      float mx = -1e30f;
#pragma unroll 8
      for (int c = 0; c < 32; ++c) {
        float s = S_lds[row][c] * SCALE;
        s = (key0 + c > q) ? -1e30f : s;
        mx = fmaxf(mx, s);
      }
      const float m_new = fmaxf(m_run, mx);
      const float alpha = __expf(m_run - m_new);
      float sum = 0.f;
#pragma unroll 8
      for (int c = 0; c < 32; ++c) {
        float s = S_lds[row][c] * SCALE;
        s = (key0 + c > q) ? -1e30f : s;
        const float p = __expf(s - m_new);
        sum += p;
        P_lds[row][c] = (__bf16)p;           // hardware cvt
      }
      l_run = l_run * alpha + sum;
      m_run = m_new;
      alpha_l[row] = alpha;
    }
    __syncthreads();

    // -- rescale O, then O += P @ V over this wave's D slice -------------
    float al[8];
#pragma unroll
    for (int r = 0; r < 8; ++r) al[r] = alpha_l[r + (hi << 3)];
#pragma unroll
    for (int f = 0; f < 8; ++f)
#pragma unroll
      for (int r = 0; r < 8; ++r) O[f][r] *= al[r];

    const v16bf aP = join88(*(const v8bf*)(&P_lds[l16][kb16]),
                            *(const v8bf*)(&P_lds[l16][kb16 + 8]));
#pragma unroll
    for (int f = 0; f < 8; ++f) {
      const __bf16* vp =
          Vt + (size_t)(b * DIM + dbase + (f << 4) + l16) * SEQ + key0 + kb16;
      const v16bf bV = *(const v16bf*)vp;
      O[f] = wmma_bf16(aP, bV, O[f]);
    }
    // No trailing barrier needed: next-iteration LDS writes are fenced by
    // the two barriers above before any wave can overwrite P_lds/alpha_l.
  }

  // -- normalize by row sums and store fp32 output -----------------------
  if (wave == 0 && lane < 16) l_l[lane] = l_run;
  __syncthreads();

  float inv[8];
#pragma unroll
  for (int r = 0; r < 8; ++r) inv[r] = 1.f / l_l[r + (hi << 3)];
#pragma unroll
  for (int f = 0; f < 8; ++f) {
    const int dcol = dbase + (f << 4) + l16;
#pragma unroll
    for (int r = 0; r < 8; ++r) {
      out[(size_t)(bS + q0 + r + (hi << 3)) * DIM + dcol] = O[f][r] * inv[r];
    }
  }
}

// ---------------------------------------------------------------------------
extern "C" void kernel_launch(void* const* d_in, const int* in_sizes, int n_in,
                              void* d_out, int out_size, void* d_ws, size_t ws_size,
                              hipStream_t stream) {
  (void)in_sizes; (void)n_in; (void)out_size; (void)ws_size;

  const float* x  = (const float*)d_in[0];
  const float* Wq = (const float*)d_in[1];
  const float* Wk = (const float*)d_in[2];
  const float* Wv = (const float*)d_in[3];
  float* out = (float*)d_out;

  // Workspace layout (bf16 elements):
  //   xb (8M) | Wb (3M) | Qb (8M) | Kb (8M) | Vt (8M)  = 35M elems = 70 MB
  const size_t NXE = (size_t)BATCH * SEQ * DIM;   // 8M
  const size_t NWE = (size_t)DIM * DIM;           // 1M
  __bf16* ws = (__bf16*)d_ws;
  __bf16* xb = ws;
  __bf16* Wb = xb + NXE;
  __bf16* Qb = Wb + 3 * NWE;
  __bf16* Kb = Qb + NXE;
  __bf16* Vt = Kb + NXE;

  cvt_bf16_kernel<<<dim3(4096, 4), 256, 0, stream>>>(
      x, Wq, Wk, Wv, xb, Wb, Wb + NWE, Wb + 2 * NWE);
  qkv_proj_kernel<<<1536, 256, 0, stream>>>(xb, Wb, Qb, Kb, Vt);
  attn_kernel<<<dim3(SEQ / 16, BATCH), 256, 0, stream>>>(Qb, Kb, Vt, out);
}
